// LightOnOCRPatchMerger_31275951849785
// MI455X (gfx1250) — matrix operand order repaired and checked by reading, not compiled, not run on hardware
//
#include <hip/hip_runtime.h>
#include <hip/hip_bf16.h>
#include <stdint.h>

// ---------------------------------------------------------------------------
// LightOnOCR patch-merger: 2x2 unfold + Linear(4096 -> 1024), fp32 in/out.
// K is permuted to k' = q*1024 + d (q = sub-patch id) so the unfold gather
// becomes contiguous 32-float token segments. Compute: bf16 hi/lo split on
// v_wmma_f32_16x16x32_bf16 (hi*hi + hi*lo + lo*hi ~= fp32 accuracy at bf16
// pipe speed). Double-buffered LDS + software pipelining hides A-tile fetch
// and conversion behind the 24 WMMAs per chunk. Conversions use native
// f32->bf16 casts (v_cvt_pk_bf16_f32) instead of manual RNE bit twiddling.
// ---------------------------------------------------------------------------

typedef __attribute__((ext_vector_type(16))) __bf16 v16bf;
typedef __attribute__((ext_vector_type(2)))  __bf16 v2bf;
typedef __attribute__((ext_vector_type(8)))  float  v8f;

#define PATCH 14
#define D_HID 1024
#define K_TOT 4096            // D * MERGE^2
#define N_OUT 1024
#define KC 32                 // k-depth per bf16 WMMA
#define NCHUNK (K_TOT / KC)   // 128 chunks
#define BM 128                // block tile M
#define BN 128                // block tile N (8 x 16-wide n-tiles)
#define NT_TOT (N_OUT / 16)   // 64 n-tiles
#define LDS_STRIDE 20         // dwords per A row in LDS (16 data + 4 pad)
#define WFRAG_DWORDS ((size_t)NT_TOT * NCHUNK * 32 * 8)  // 2,097,152 dwords

union FragBF { v16bf v; uint4 q[2]; };
union PairBF { v2bf v; uint32_t u; };

// Pack two floats into one dword of bf16 (RNE via native fptrunc).
__device__ __forceinline__ uint32_t pack_bf16(float a, float b) {
    PairBF p;
    p.v = (v2bf){(__bf16)a, (__bf16)b};
    return p.u;
}

// hi = bf16(x); lo = bf16(x - float(hi)); packed pairwise.
__device__ __forceinline__ void split_pair(float a, float b,
                                           uint32_t& hp, uint32_t& lp) {
    __bf16 ha = (__bf16)a, hb = (__bf16)b;
    float ra = a - (float)ha, rb = b - (float)hb;   // bf16->f32 is exact
    PairBF h, l;
    h.v = (v2bf){ha, hb};
    l.v = (v2bf){(__bf16)ra, (__bf16)rb};
    hp = h.u;
    lp = l.u;
}

// ---------------------------------------------------------------------------
// One-shot: permute weight [N, d*4+q] -> k' = q*1024+d, split into bf16 hi/lo,
// stored as WMMA B fragments (32x16 bf16) in contiguous k order, same as A:
//   lanes 0-15: N = lane,     VGPR j = K {2j, 2j+1}       (K in [0,16))
//   lanes 16-31: N = lane-16, VGPR j = K {16+2j, 17+2j}   (K in [16,32))
// dword index = ((nt*NCHUNK + kc)*32 + lane)*8 + j
// ---------------------------------------------------------------------------
__global__ __launch_bounds__(256)
void pack_w_kernel(const float* __restrict__ w,
                   uint32_t* __restrict__ whi, uint32_t* __restrict__ wlo) {
    uint32_t t = blockIdx.x * 256u + threadIdx.x;   // [0, WFRAG_DWORDS)
    int j    = t & 7;
    int lane = (t >> 3) & 31;
    int kc   = (t >> 8) & (NCHUNK - 1);
    int nt   = t >> 15;
    int n    = nt * 16 + (lane & 15);
    int k0   = kc * KC + ((lane >> 4) << 4) + 2 * j;

    int q0 = k0 >> 10, d0 = k0 & 1023;               // k' = q*1024 + d
    int q1 = (k0 + 1) >> 10, d1 = (k0 + 1) & 1023;
    float x0 = w[(size_t)n * K_TOT + d0 * 4 + q0];   // weight[n][d*4+q]
    float x1 = w[(size_t)n * K_TOT + d1 * 4 + q1];
    uint32_t hp, lp;
    split_pair(x0, x1, hp, lp);
    whi[t] = hp;
    wlo[t] = lp;
}

// ---------------------------------------------------------------------------
// GEMM: C[M,1024] = A'[M,4096] x W'^T, A' gathered on the fly from tokens.
// ---------------------------------------------------------------------------
__global__ __launch_bounds__(256)
void merger_gemm_kernel(const float* __restrict__ feat,
                        const long long* __restrict__ sizes,
                        const uint32_t* __restrict__ whi,
                        const uint32_t* __restrict__ wlo,
                        float* __restrict__ out, int M, int n_img) {
    __shared__ __align__(16) uint32_t sAhi[2][BM * LDS_STRIDE];
    __shared__ __align__(16) uint32_t sAlo[2][BM * LDS_STRIDE];
    __shared__ int sTok[BM][4];

    const int tid    = threadIdx.x;
    const int lane   = tid & 31;
    const int wave   = tid >> 5;
    const int waveM  = wave & 3;   // 4 waves along M (32 rows each)
    const int waveN  = wave >> 2;  // 2 waves along N (64 cols each)
    const int half   = lane >> 4;
    const int lane16 = lane & 15;
    const int m0     = blockIdx.y * BM;

    // Per-row token table: merged row -> the 4 token rows (q = kh*2+kw order).
    if (tid < BM) {
        int m = m0 + tid;
        int t0 = -1, t2 = -1;
        long long tokOff = 0, merOff = 0;
        for (int i = 0; i < n_img; ++i) {
            int h = (int)(sizes[2 * i] / PATCH);
            int w = (int)(sizes[2 * i + 1] / PATCH);
            int hm = h >> 1, wm = w >> 1;
            long long nm = (long long)hm * wm;
            if (m >= merOff && m < merOff + nm) {
                long long loc = m - merOff;
                int rb = (int)(loc / wm), cb = (int)(loc % wm);
                t0 = (int)(tokOff + (long long)(2 * rb) * w + 2 * cb);
                t2 = t0 + w;
            }
            tokOff += (long long)h * w;
            merOff += nm;
        }
        sTok[tid][0] = t0; sTok[tid][1] = (t0 < 0) ? -1 : t0 + 1;
        sTok[tid][2] = t2; sTok[tid][3] = (t2 < 0) ? -1 : t2 + 1;
    }
    __syncthreads();   // token table ready for the pipeline prologue

    v8f acc[2][4];
#pragma unroll
    for (int i = 0; i < 2; ++i)
#pragma unroll
        for (int j = 0; j < 4; ++j)
            acc[i][j] = (v8f){0.f, 0.f, 0.f, 0.f, 0.f, 0.f, 0.f, 0.f};

    float4 x[4];
    // ---- pipeline prologue: fetch + stage chunk 0 into buffer 0 ----------
    {
#pragma unroll
        for (int i = 0; i < 4; ++i) {
            int id = tid + i * 256, row = id >> 3, seg = id & 7;
            int tok = sTok[row][0];
            x[i] = make_float4(0.f, 0.f, 0.f, 0.f);
            if (tok >= 0)
                x[i] = *reinterpret_cast<const float4*>(
                           feat + (size_t)tok * D_HID + seg * 4);
        }
#pragma unroll
        for (int i = 0; i < 4; ++i) {
            int id = tid + i * 256, row = id >> 3, seg = id & 7;
            uint32_t h0, h1, l0, l1;
            split_pair(x[i].x, x[i].y, h0, l0);
            split_pair(x[i].z, x[i].w, h1, l1);
            uint32_t* ph = &sAhi[0][row * LDS_STRIDE + seg * 2];
            uint32_t* pl = &sAlo[0][row * LDS_STRIDE + seg * 2];
            ph[0] = h0; ph[1] = h1;
            pl[0] = l0; pl[1] = l1;
        }
    }

    for (int kc = 0; kc < NCHUNK; ++kc) {
        const int cur = kc & 1, nxt = cur ^ 1;
        const bool hasNext = (kc + 1) < NCHUNK;
        __syncthreads();   // buf[cur] writes visible; buf[nxt] readers done

        // ---- issue next chunk's global loads early (latency hidden) ------
        if (hasNext) {
            const int qn = (kc + 1) >> 5;
            const int dn = ((kc + 1) & 31) * KC;
#pragma unroll
            for (int i = 0; i < 4; ++i) {
                int id = tid + i * 256, row = id >> 3, seg = id & 7;
                int tok = sTok[row][qn];
                x[i] = make_float4(0.f, 0.f, 0.f, 0.f);
                if (tok >= 0)
                    x[i] = *reinterpret_cast<const float4*>(
                               feat + (size_t)tok * D_HID + dn + seg * 4);
            }
        }

        // ---- A fragments from LDS (16x32 bf16 operand layout) ------------
        FragBF aHi[2], aLo[2];
#pragma unroll
        for (int mm = 0; mm < 2; ++mm) {
            int row = waveM * 32 + mm * 16 + lane16;
            const uint32_t* bh = &sAhi[cur][row * LDS_STRIDE + half * 8];
            const uint32_t* bl = &sAlo[cur][row * LDS_STRIDE + half * 8];
            aHi[mm].q[0] = *reinterpret_cast<const uint4*>(bh);
            aHi[mm].q[1] = *reinterpret_cast<const uint4*>(bh + 4);
            aLo[mm].q[0] = *reinterpret_cast<const uint4*>(bl);
            aLo[mm].q[1] = *reinterpret_cast<const uint4*>(bl + 4);
        }

        // ---- all 8 B fragments up front (one clause, pipelined waits) ----
        FragBF bHi[4], bLo[4];
#pragma unroll
        for (int nn = 0; nn < 4; ++nn) {
            int nt = blockIdx.x * 8 + waveN * 4 + nn;
            size_t off = ((size_t)(nt * NCHUNK + kc) * 32 + lane) * 8;
            const uint4* ph = reinterpret_cast<const uint4*>(whi + off);
            const uint4* pl = reinterpret_cast<const uint4*>(wlo + off);
            bHi[nn].q[0] = ph[0]; bHi[nn].q[1] = ph[1];
            bLo[nn].q[0] = pl[0]; bLo[nn].q[1] = pl[1];
        }

        // ---- 24 WMMAs in 3 passes: dependent pairs are 8 instrs apart ----
#pragma unroll
        for (int nn = 0; nn < 4; ++nn)
#pragma unroll
            for (int mm = 0; mm < 2; ++mm)
                acc[mm][nn] = __builtin_amdgcn_wmma_f32_16x16x32_bf16(
                    false, aHi[mm].v, false, bHi[nn].v, (short)0, acc[mm][nn], false, false);
#pragma unroll
        for (int nn = 0; nn < 4; ++nn)
#pragma unroll
            for (int mm = 0; mm < 2; ++mm)
                acc[mm][nn] = __builtin_amdgcn_wmma_f32_16x16x32_bf16(
                    false, aHi[mm].v, false, bLo[nn].v, (short)0, acc[mm][nn], false, false);
#pragma unroll
        for (int nn = 0; nn < 4; ++nn)
#pragma unroll
            for (int mm = 0; mm < 2; ++mm)
                acc[mm][nn] = __builtin_amdgcn_wmma_f32_16x16x32_bf16(
                    false, aLo[mm].v, false, bHi[nn].v, (short)0, acc[mm][nn], false, false);

        // ---- convert + stage next chunk into the other buffer ------------
        if (hasNext) {
#pragma unroll
            for (int i = 0; i < 4; ++i) {
                int id = tid + i * 256, row = id >> 3, seg = id & 7;
                uint32_t h0, h1, l0, l1;
                split_pair(x[i].x, x[i].y, h0, l0);
                split_pair(x[i].z, x[i].w, h1, l1);
                uint32_t* ph = &sAhi[nxt][row * LDS_STRIDE + seg * 2];
                uint32_t* pl = &sAlo[nxt][row * LDS_STRIDE + seg * 2];
                ph[0] = h0; ph[1] = h1;
                pl[0] = l0; pl[1] = l1;
            }
        }
    }

    // ---- Store C: VGPR v, lanes<16 -> row v, lanes>=16 -> row v+8 --------
#pragma unroll
    for (int mm = 0; mm < 2; ++mm) {
        int rowBase = m0 + waveM * 32 + mm * 16 + half * 8;
#pragma unroll
        for (int nn = 0; nn < 4; ++nn) {
            int col = (blockIdx.x * 8 + waveN * 4 + nn) * 16 + lane16;
#pragma unroll
            for (int v = 0; v < 8; ++v) {
                int r = rowBase + v;
                if (r < M) out[(size_t)r * N_OUT + col] = acc[mm][nn][v];
            }
        }
    }
}

extern "C" void kernel_launch(void* const* d_in, const int* in_sizes, int n_in,
                              void* d_out, int out_size, void* d_ws, size_t ws_size,
                              hipStream_t stream) {
    const float*     feat  = (const float*)d_in[0];
    const float*     w     = (const float*)d_in[1];
    const long long* sizes = (const long long*)d_in[2];
    float*           out   = (float*)d_out;

    const int n_img = in_sizes[2] / 2;
    const int M     = out_size / N_OUT;          // 11955 merged rows

    // ws: bf16 hi/lo weight fragments, 2 * 8 MB.
    uint32_t* whi = (uint32_t*)d_ws;
    uint32_t* wlo = whi + WFRAG_DWORDS;

    pack_w_kernel<<<(unsigned)(WFRAG_DWORDS / 256), 256, 0, stream>>>(w, whi, wlo);

    dim3 grid(N_OUT / BN, (M + BM - 1) / BM);    // N fastest -> A tiles hit L2
    merger_gemm_kernel<<<grid, 256, 0, stream>>>(feat, sizes, whi, wlo, out, M, n_img);
}
